// MoE_29712583753914
// MI455X (gfx1250) — compile-verified, matrix operand order checked by reading, not run on hardware
//
#include <hip/hip_runtime.h>
#include <hip/hip_bf16.h>

#ifndef __has_builtin
#define __has_builtin(x) 0
#endif

// ---------------- problem constants ----------------
#define NTOK   8192      // B*T
#define DIM    512       // D
#define HID    512       // H
#define NEXP   8         // E
#define CAP    NTOK      // per-expert bucket capacity (token appears <=1x per expert)
#define LDS_STRIDE 520   // 512 + 8 halves pad (1040B row; 16B aligned, bank-friendly)

typedef __attribute__((ext_vector_type(16))) __bf16 v16bf;
typedef __attribute__((ext_vector_type(8)))  float  v8f;
typedef unsigned int v4u __attribute__((ext_vector_type(4)));
typedef int          v8i __attribute__((ext_vector_type(8)));
typedef int          v4i __attribute__((ext_vector_type(4)));

union FragBF {
    uint4 q[2];
    v16bf v;
};

static __device__ __forceinline__ unsigned short f32_to_bf16(float f) {
    unsigned int u = __float_as_uint(f);
    u += 0x7FFFu + ((u >> 16) & 1u);       // round-to-nearest-even
    return (unsigned short)(u >> 16);
}

// ---------------------------------------------------------------------------
// x fp32 -> bf16 (feeds the TDM gather in the expert kernel)
// ---------------------------------------------------------------------------
__global__ __launch_bounds__(256) void moe_xconv_kernel(
    const float* __restrict__ x, unsigned short* __restrict__ xbf)
{
    int i = blockIdx.x * blockDim.x + threadIdx.x;      // one float4 per thread
    float4 v = ((const float4*)x)[i];
    ushort4 o;
    o.x = f32_to_bf16(v.x); o.y = f32_to_bf16(v.y);
    o.z = f32_to_bf16(v.z); o.w = f32_to_bf16(v.w);
    ((ushort4*)xbf)[i] = o;
}

// ---------------------------------------------------------------------------
// Gate: logits = x @ Wg + bg, top-2 + softmax, scatter into per-expert buckets
// one wave32 per token
// ---------------------------------------------------------------------------
__global__ __launch_bounds__(256) void moe_gate_kernel(
    const float* __restrict__ x, const float* __restrict__ Wg,
    const float* __restrict__ bg,
    int* __restrict__ cnt, int* __restrict__ btok, float* __restrict__ bw)
{
    int tok  = (blockIdx.x * blockDim.x + threadIdx.x) >> 5;
    int lane = threadIdx.x & 31;
    if (tok >= NTOK) return;

    const float* xr = x + (size_t)tok * DIM;
    float acc[NEXP];
#pragma unroll
    for (int e = 0; e < NEXP; ++e) acc[e] = 0.f;

    for (int d = lane; d < DIM; d += 32) {
        float xv = xr[d];
        const float* wr = Wg + (size_t)d * NEXP;
#pragma unroll
        for (int e = 0; e < NEXP; ++e) acc[e] += xv * wr[e];
    }
#pragma unroll
    for (int e = 0; e < NEXP; ++e) {
        float v = acc[e];
#pragma unroll
        for (int off = 16; off > 0; off >>= 1) v += __shfl_xor(v, off, 32);
        acc[e] = v;
    }
    if (lane == 0) {
        float lg[NEXP];
#pragma unroll
        for (int e = 0; e < NEXP; ++e) lg[e] = acc[e] + bg[e];
        int e0 = 0; float s0 = lg[0];
#pragma unroll
        for (int e = 1; e < NEXP; ++e) if (lg[e] > s0) { s0 = lg[e]; e0 = e; }
        int e1 = (e0 == 0) ? 1 : 0; float s1 = lg[e1];
#pragma unroll
        for (int e = 0; e < NEXP; ++e)
            if (e != e0 && lg[e] > s1) { s1 = lg[e]; e1 = e; }
        float w0 = 1.0f / (1.0f + __expf(s1 - s0));   // softmax over 2
        float w1 = 1.0f - w0;
        int p0 = atomicAdd(&cnt[e0], 1);
        btok[e0 * CAP + p0] = tok;  bw[e0 * CAP + p0] = w0;
        int p1 = atomicAdd(&cnt[e1], 1);
        btok[e1 * CAP + p1] = tok;  bw[e1 * CAP + p1] = w1;
    }
}

// ---------------------------------------------------------------------------
// Weight prep: fp32 [E][R][C] -> bf16 transposed [E][C][R]  (R=C=512)
// So B-matrix WMMA fragments (16 contiguous K elems of one output column)
// become contiguous 32-byte loads.
// ---------------------------------------------------------------------------
__global__ __launch_bounds__(256) void moe_transpose_bf16_kernel(
    const float* __restrict__ src, unsigned short* __restrict__ dst)
{
    __shared__ unsigned short tile[32][33];
    int e  = blockIdx.z;
    int r0 = blockIdx.y * 32;      // source row block (K dim)
    int c0 = blockIdx.x * 32;      // source col block (N dim)
    int c  = threadIdx.x & 31;
    int rb = threadIdx.x >> 5;     // 0..7
    const float*     s = src + (size_t)e * DIM * HID;
    unsigned short*  d = dst + (size_t)e * DIM * HID;
#pragma unroll
    for (int i = 0; i < 4; ++i) {
        int r = rb + i * 8;
        tile[r][c] = f32_to_bf16(s[(size_t)(r0 + r) * HID + c0 + c]);
    }
    __syncthreads();
#pragma unroll
    for (int i = 0; i < 4; ++i) {
        int r = rb + i * 8;        // along transposed rows (source cols)
        d[(size_t)(c0 + r) * DIM + r0 + c] = tile[c][r];
    }
}

// ---------------------------------------------------------------------------
// Grouped expert GEMM: each block = one 16-token tile of one expert.
//   TDM gather: 16 token rows of xbf DMA'd into LDS (hardware pad to 520h rows)
//   h = gelu(x_tile @ W1[e] + b1[e])
//   y = h @ W2[e] + b2[e]
//   out[tok] += gate_w * y   (float atomics; out pre-zeroed)
// ---------------------------------------------------------------------------
__global__ __launch_bounds__(128) void moe_expert_kernel(
    const unsigned short* __restrict__ xbf,
    const float* __restrict__ b1, const float* __restrict__ b2,
    const unsigned short* __restrict__ w1t, const unsigned short* __restrict__ w2t,
    const int* __restrict__ cnt, const int* __restrict__ btok,
    const float* __restrict__ bw, float* __restrict__ out)
{
    __shared__ __align__(16) unsigned short sx[16 * LDS_STRIDE];
    __shared__ __align__(16) unsigned short sh[16 * LDS_STRIDE];
    __shared__ int   stok[16];
    __shared__ float swt[16];
    __shared__ int   sinfo[2];

    int tid = threadIdx.x;

    // --- map block -> (expert, tile start) via prefix over bucket counts ---
    if (tid == 0) {
        int t = blockIdx.x, e = -1, start = 0;
        for (int i = 0; i < NEXP; ++i) {
            int tiles = (cnt[i] + 15) >> 4;
            if (t < tiles) { e = i; start = t * 16; break; }
            t -= tiles;
        }
        sinfo[0] = e; sinfo[1] = start;
    }
    __syncthreads();
    int e = sinfo[0];
    if (e < 0) return;                 // uniform branch: surplus block
    int start = sinfo[1];
    int c_e   = cnt[e];

    if (tid < 16) {
        int idx = start + tid;
        int tok = 0; float w = 0.f;    // pad rows: weight 0 -> contributes 0
        if (idx < c_e) { tok = btok[e * CAP + idx]; w = bw[e * CAP + idx]; }
        stok[tid] = tok; swt[tid] = w;
    }

    const int lane = tid & 31;
    const int wave = tid >> 5;

#if __has_builtin(__builtin_amdgcn_tensor_load_to_lds)
    // --- Tensor Data Mover gather: 16 token rows (bf16) -> LDS, padded rows ---
    if (wave == 0) {
        unsigned idxw[8];
#pragma unroll
        for (int j = 0; j < 8; ++j) {                       // 16-bit gather indices
            int i0 = start + 2 * j, i1 = i0 + 1;
            unsigned t0 = (i0 < c_e) ? (unsigned)btok[e * CAP + i0] : 0u;
            unsigned t1 = (i1 < c_e) ? (unsigned)btok[e * CAP + i1] : 0u;
            idxw[j] = __builtin_amdgcn_readfirstlane(t0 | (t1 << 16));
        }
        unsigned long long ga = (unsigned long long)(uintptr_t)xbf;
        unsigned lds_off = (unsigned)(uintptr_t)(&sx[0]);   // low 32b = LDS offset

        v4u g0;
        g0[0] = 1u | (1u << 31);                            // count=1, gather_mode, 16b idx
        g0[1] = __builtin_amdgcn_readfirstlane(lds_off);
        g0[2] = __builtin_amdgcn_readfirstlane((unsigned)ga);
        g0[3] = __builtin_amdgcn_readfirstlane(
                    ((unsigned)(ga >> 32) & 0x01FFFFFFu) | (2u << 30));  // type=2
        v8i g1;
        g1[0] = (int)((1u << 16)            // data_size = 2B
                    | (1u << 20)            // pad_enable
                    | (7u << 22)            // pad_interval: 1024B
                    | (3u << 25));          // pad_amount: 16B  -> 520-half rows
        g1[1] = (int)(512u << 16);          // tensor_dim0 = 512
        g1[2] = (int)((unsigned)NTOK << 16);// tensor_dim1 = 8192
        g1[3] = (int)(512u << 16);          // tile_dim0 = 512
        g1[4] = 16;                         // tile_dim1 = 16 gather indices
        g1[5] = 512;                        // tensor_dim0_stride
        g1[6] = 0; g1[7] = 0;
        v4i g2, g3;
#pragma unroll
        for (int j = 0; j < 4; ++j) { g2[j] = (int)idxw[j]; g3[j] = (int)idxw[4 + j]; }

#if __has_include(<hip/amd_detail/amd_gfx1250_TDM.h>)
        v8i g4 = {0, 0, 0, 0, 0, 0, 0, 0};
        __builtin_amdgcn_tensor_load_to_lds(g0, g1, g2, g3, g4, 0);   // clang-23 form
#else
        __builtin_amdgcn_tensor_load_to_lds(g0, g1, g2, g3, 0);       // ROCm 7.2 form
#endif
        __builtin_amdgcn_s_wait_tensorcnt(0);
    }
    __syncthreads();
#else
    __syncthreads();
    for (int i = tid; i < 16 * DIM; i += 128) {
        int r = i >> 9, col = i & (DIM - 1);
        sx[r * LDS_STRIDE + col] = xbf[(size_t)stok[r] * DIM + col];
    }
    __syncthreads();
#endif

    const int nlo  = lane & 15;          // N column within tile / A row
    const int khi  = (lane >> 4) << 4;   // 0 or 16: contiguous K half (k = 16h + s), same for A and B

    // ================= stage 1: h = gelu(x @ W1 + b1) =================
    for (int nt = wave; nt < HID / 16; nt += 4) {
        int nb = nt * 16;
        const unsigned short* brow =
            w1t + ((size_t)(e * HID + nb + nlo)) * DIM + khi;
        __builtin_prefetch(brow, 0, 1);
        v8f acc = {};
#pragma unroll
        for (int kb = 0; kb < DIM / 32; ++kb) {
            FragBF a, b;
            const uint4* ap = (const uint4*)(sx + nlo * LDS_STRIDE + kb * 32 + khi);
            a.q[0] = ap[0]; a.q[1] = ap[1];
            const uint4* bp = (const uint4*)(brow + (size_t)kb * 32);
            b.q[0] = bp[0]; b.q[1] = bp[1];
            acc = __builtin_amdgcn_wmma_f32_16x16x32_bf16(
                false, a.v, false, b.v, (short)0, acc, false, false);
        }
        float bias = b1[e * HID + nb + nlo];
#pragma unroll
        for (int i = 0; i < 8; ++i) {
            float v = acc[i] + bias;
            float g = 0.5f * v * (1.0f + erff(v * 0.70710678f));   // exact gelu
            int row = i + ((lane >> 4) << 3);
            sh[row * LDS_STRIDE + nb + nlo] = f32_to_bf16(g);
        }
    }
    __syncthreads();

    // ================= stage 2: y = h @ W2 + b2, combine =================
    for (int nt = wave; nt < HID / 16; nt += 4) {
        int nb = nt * 16;
        const unsigned short* brow =
            w2t + ((size_t)(e * HID + nb + nlo)) * HID + khi;
        __builtin_prefetch(brow, 0, 1);
        v8f acc = {};
#pragma unroll
        for (int kb = 0; kb < HID / 32; ++kb) {
            FragBF a, b;
            const uint4* ap = (const uint4*)(sh + nlo * LDS_STRIDE + kb * 32 + khi);
            a.q[0] = ap[0]; a.q[1] = ap[1];
            const uint4* bp = (const uint4*)(brow + (size_t)kb * 32);
            b.q[0] = bp[0]; b.q[1] = bp[1];
            acc = __builtin_amdgcn_wmma_f32_16x16x32_bf16(
                false, a.v, false, b.v, (short)0, acc, false, false);
        }
        float bias = b2[e * HID + nb + nlo];
#pragma unroll
        for (int i = 0; i < 8; ++i) {
            int row = i + ((lane >> 4) << 3);
            float v = acc[i] + bias;
            atomicAdd(&out[(size_t)stok[row] * HID + nb + nlo], swt[row] * v);
        }
    }
}

// ---------------------------------------------------------------------------
extern "C" void kernel_launch(void* const* d_in, const int* in_sizes, int n_in,
                              void* d_out, int out_size, void* d_ws, size_t ws_size,
                              hipStream_t stream) {
    const float* x  = (const float*)d_in[0];
    const float* Wg = (const float*)d_in[1];
    const float* bg = (const float*)d_in[2];
    const float* W1 = (const float*)d_in[3];
    const float* b1 = (const float*)d_in[4];
    const float* W2 = (const float*)d_in[5];
    const float* b2 = (const float*)d_in[6];
    float* out = (float*)d_out;

    char* ws = (char*)d_ws;
    int*            cnt  = (int*)ws;                                   // 8 ints
    int*            btok = (int*)(ws + 256);                           // E*CAP ints
    float*          bw   = (float*)(ws + 256 + (size_t)NEXP * CAP * 4);
    unsigned short* w1t  = (unsigned short*)(ws + 256 + (size_t)NEXP * CAP * 8);
    unsigned short* w2t  = w1t + (size_t)NEXP * DIM * HID;
    unsigned short* xbf  = w2t + (size_t)NEXP * DIM * HID;

    hipMemsetAsync(cnt, 0, NEXP * sizeof(int), stream);
    hipMemsetAsync(out, 0, (size_t)out_size * sizeof(float), stream);

    moe_xconv_kernel<<<(NTOK * DIM / 4) / 256, 256, 0, stream>>>(x, xbf);
    moe_gate_kernel<<<NTOK / 8, 256, 0, stream>>>(x, Wg, bg, cnt, btok, bw);

    dim3 tgrid(HID / 32, DIM / 32, NEXP);
    moe_transpose_bf16_kernel<<<tgrid, 256, 0, stream>>>(W1, w1t);
    moe_transpose_bf16_kernel<<<tgrid, 256, 0, stream>>>(W2, w2t);

    // max tiles = sum ceil(cnt_e/16) <= 2*N/16 + E
    int max_tiles = (2 * NTOK) / 16 + NEXP;
    moe_expert_kernel<<<max_tiles, 128, 0, stream>>>(
        xbf, b1, b2, w1t, w2t, cnt, btok, bw, out);
}
